// HGRNBitAttention_1597727834558
// MI455X (gfx1250) — 8-bit matrix operand order repaired and checked by reading, compiled, not run on hardware
//
#include <hip/hip_runtime.h>
#include <hip/hip_bf16.h>


typedef __attribute__((ext_vector_type(8))) int v8i;
typedef __attribute__((ext_vector_type(4))) int v4i;
typedef signed char i8;

#define B_    2
#define T_    2048
#define HID_  2048
#define ROWS_ (B_ * T_)
#define WELEMS (HID_ * HID_)
#define NCH   (B_ * HID_)
#define CHUNKS 16
#define CLEN  (T_ / CHUNKS)

__device__ __forceinline__ float sigmf(float x) { return 1.0f / (1.0f + __expf(-x)); }

__device__ __forceinline__ float block_sum(float v, float* red) {
  int tid = threadIdx.x;
  red[tid] = v; __syncthreads();
  #pragma unroll
  for (int s = 128; s > 0; s >>= 1) {
    if (tid < s) red[tid] += red[tid + s];
    __syncthreads();
  }
  float r = red[0]; __syncthreads();
  return r;
}

__device__ __forceinline__ float block_max(float v, float* red) {
  int tid = threadIdx.x;
  red[tid] = v; __syncthreads();
  #pragma unroll
  for (int s = 128; s > 0; s >>= 1) {
    if (tid < s) red[tid] = fmaxf(red[tid], red[tid + s]);
    __syncthreads();
  }
  float r = red[0]; __syncthreads();
  return r;
}


__global__ __launch_bounds__(256) void wabs_partial(const float* __restrict__ w,
                                                    float* __restrict__ part) {
  __shared__ float red[256];
  int tid = threadIdx.x;
  size_t base = (size_t)blockIdx.x * 4096;
  float s = 0.0f;
  #pragma unroll
  for (int i = 0; i < 16; ++i) s += fabsf(w[base + tid + i * 256]);
  float t = block_sum(s, red);
  if (tid == 0) part[blockIdx.x] = t;
}

__global__ __launch_bounds__(256) void wabs_final(const float* __restrict__ part,
                                                  float* __restrict__ wscale) {
  __shared__ float red[256];
  int tid = threadIdx.x;
  float s = part[tid] + part[tid + 256] + part[tid + 512] + part[tid + 768];
  float t = block_sum(s, red);
  if (tid == 0) wscale[0] = fmaxf(t * (1.0f / (float)WELEMS), 1e-5f);
}

__global__ __launch_bounds__(256) void wquant(const float* __restrict__ w,
                                              const float* __restrict__ wscale,
                                              i8* __restrict__ q) {
  float inv = 1.0f / wscale[0];
  size_t i = ((size_t)blockIdx.x * 256 + threadIdx.x) * 4;
  #pragma unroll
  for (int j = 0; j < 4; ++j) {
    float v = rintf(w[i + j] * inv);
    v = fminf(fmaxf(v, -1.0f), 1.0f);
    q[i + j] = (i8)v;
  }
}


__global__ __launch_bounds__(256) void act_rms_quant3(
    const float* __restrict__ hs,
    const float* __restrict__ nI, const float* __restrict__ nF, const float* __restrict__ nG,
    i8* __restrict__ qI, i8* __restrict__ qF, i8* __restrict__ qG,
    float* __restrict__ sI, float* __restrict__ sF, float* __restrict__ sG) {
  __shared__ float red[256];
  const int tid = threadIdx.x;
  const size_t row = blockIdx.x;
  const float* x = hs + row * (size_t)HID_;
  float v[8];
  float ss = 0.0f;
  #pragma unroll
  for (int j = 0; j < 8; ++j) { v[j] = x[j * 256 + tid]; ss += v[j] * v[j]; }
  float tot = block_sum(ss, red);
  float rinv = rsqrtf(tot * (1.0f / (float)HID_) + 1e-8f);
  #pragma unroll
  for (int j = 0; j < 8; ++j) v[j] *= rinv;

#define DO_PROJ(NW, Q, S)                                                      \
  {                                                                            \
    float y[8]; float am = 0.0f;                                               \
    _Pragma("unroll")                                                          \
    for (int j = 0; j < 8; ++j) {                                              \
      y[j] = v[j] * NW[j * 256 + tid];                                         \
      am = fmaxf(am, fabsf(y[j]));                                             \
    }                                                                          \
    float amax = fmaxf(block_max(am, red), 1e-5f);                             \
    float s = 127.0f / amax;                                                   \
    _Pragma("unroll")                                                          \
    for (int j = 0; j < 8; ++j) {                                              \
      float q = fminf(fmaxf(rintf(y[j] * s), -128.0f), 127.0f);                \
      Q[row * (size_t)HID_ + j * 256 + tid] = (i8)q;                           \
    }                                                                          \
    if (tid == 0) S[row] = amax * (1.0f / 127.0f);                             \
  }

  DO_PROJ(nI, qI, sI)
  DO_PROJ(nF, qF, sF)
  DO_PROJ(nG, qG, sG)
}


#define BM 128
#define BN 128
#define BKK 64
#define LDT 80

__global__ __launch_bounds__(256) void gemm_iu8(
    const i8* __restrict__ A, const i8* __restrict__ W,
    const float* __restrict__ aScale, const float* __restrict__ wScale,
    float* __restrict__ C, int M, int N, int K) {
  __shared__ alignas(16) i8 As[2][BM][LDT];
  __shared__ alignas(16) i8 Ws[2][BN][LDT];

  const int tid = threadIdx.x;
  const int lane = tid & 31;
  const int wave = tid >> 5;
  const int wm0 = (wave & 3) * 32;
  const int wn0 = (wave >> 2) * 64;
  const int mBase = blockIdx.y * BM;
  const int nBase = blockIdx.x * BN;
  const int mfr = lane & 15;
  const int half = lane >> 4;

  const int sr0 = tid >> 2;
  const int sr1 = sr0 + 64;
  const int sko = (tid & 3) << 4;
  const i8* aRow0 = A + (size_t)(mBase + sr0) * K + sko;
  const i8* aRow1 = A + (size_t)(mBase + sr1) * K + sko;
  const i8* wRow0 = W + (size_t)(nBase + sr0) * K + sko;
  const i8* wRow1 = W + (size_t)(nBase + sr1) * K + sko;

  v8i acc[2][4];
  const v8i vz = {0, 0, 0, 0, 0, 0, 0, 0};
  #pragma unroll
  for (int i = 0; i < 2; ++i)
    #pragma unroll
    for (int j = 0; j < 4; ++j) acc[i][j] = vz;

  {
    v4i a0 = *(const v4i*)(aRow0);
    v4i a1 = *(const v4i*)(aRow1);
    v4i w0 = *(const v4i*)(wRow0);
    v4i w1 = *(const v4i*)(wRow1);
    *(v4i*)&As[0][sr0][sko] = a0;
    *(v4i*)&As[0][sr1][sko] = a1;
    *(v4i*)&Ws[0][sr0][sko] = w0;
    *(v4i*)&Ws[0][sr1][sko] = w1;
  }
  __syncthreads();

  int buf = 0;
  for (int k0 = 0; k0 < K; k0 += BKK, buf ^= 1) {
    const bool hasNext = (k0 + BKK) < K;
    v4i na0, na1, nw0, nw1;
    if (hasNext) {
      na0 = *(const v4i*)(aRow0 + k0 + BKK);
      na1 = *(const v4i*)(aRow1 + k0 + BKK);
      nw0 = *(const v4i*)(wRow0 + k0 + BKK);
      nw1 = *(const v4i*)(wRow1 + k0 + BKK);
    }

    v8i af[2];
    #pragma unroll
    for (int mt = 0; mt < 2; ++mt) {
      const i8* rp = &As[buf][wm0 + mt * 16 + mfr][half << 4];
      #pragma unroll
      for (int p = 0; p < 4; ++p) {
        const int* s2 = (const int*)(rp + ((p >> 1) << 5) + ((p & 1) << 3));
        af[mt][2 * p]     = s2[0];
        af[mt][2 * p + 1] = s2[1];
      }
    }
    v8i bf[4];
    #pragma unroll
    for (int nt = 0; nt < 4; ++nt) {
      const i8* rp = &Ws[buf][wn0 + nt * 16 + mfr][half << 4];
      v4i g0 = *(const v4i*)(rp);
      v4i g1 = *(const v4i*)(rp + 32);
      bf[nt][0] = g0[0]; bf[nt][1] = g0[1]; bf[nt][2] = g0[2]; bf[nt][3] = g0[3];
      bf[nt][4] = g1[0]; bf[nt][5] = g1[1]; bf[nt][6] = g1[2]; bf[nt][7] = g1[3];
    }

    #pragma unroll
    for (int mt = 0; mt < 2; ++mt)
      #pragma unroll
      for (int nt = 0; nt < 4; ++nt)
        acc[mt][nt] = __builtin_amdgcn_wmma_i32_16x16x64_iu8(
            true, af[mt], true, bf[nt], acc[mt][nt], false, false);

    if (hasNext) {
      int nb = buf ^ 1;
      *(v4i*)&As[nb][sr0][sko] = na0;
      *(v4i*)&As[nb][sr1][sko] = na1;
      *(v4i*)&Ws[nb][sr0][sko] = nw0;
      *(v4i*)&Ws[nb][sr1][sko] = nw1;
    }
    __syncthreads();
  }

  const float wsc = wScale[0];
  #pragma unroll
  for (int mt = 0; mt < 2; ++mt) {
    int mrow = mBase + wm0 + mt * 16 + half * 8;
    float sc[8];
    #pragma unroll
    for (int r = 0; r < 8; ++r) sc[r] = aScale[mrow + r] * wsc;
    #pragma unroll
    for (int nt = 0; nt < 4; ++nt) {
      int ncol = nBase + wn0 + nt * 16 + mfr;
      #pragma unroll
      for (int r = 0; r < 8; ++r)
        C[(size_t)(mrow + r) * N + ncol] = (float)acc[mt][nt][r] * sc[r];
    }
  }
}


__global__ __launch_bounds__(256) void gates_ew(float* __restrict__ li,
                                                float* __restrict__ lf, size_t n) {
  size_t i = (size_t)blockIdx.x * 256 + threadIdx.x;
  if (i >= n) return;
  float f = sigmf(lf[i]);
  float x = li[i];
  lf[i] = f;
  li[i] = x * sigmf(x) * (1.0f - f);
}


__global__ __launch_bounds__(256) void scanA(float* __restrict__ li, float* __restrict__ lf,
                                             float* __restrict__ cF, float* __restrict__ cH) {
  int gid = blockIdx.x * 256 + threadIdx.x;
  int j = gid >> 12;
  int ch = gid & 4095;
  int b = ch >> 11;
  int hid = ch & 2047;
  size_t base = ((size_t)b * T_ + (size_t)j * CLEN) * HID_ + hid;
  float h = 0.0f, P = 1.0f;
  for (int t = 0; t < CLEN; ++t) {
    size_t idx = base + (size_t)t * HID_;
    float f = lf[idx], iv = li[idx];
    h = f * h + iv;
    P *= f;
    li[idx] = h;
    lf[idx] = P;
  }
  cF[j * NCH + ch] = P;
  cH[j * NCH + ch] = h;
}

__global__ __launch_bounds__(256) void scanB(const float* __restrict__ cF,
                                             const float* __restrict__ cH,
                                             float* __restrict__ cin) {
  int ch = blockIdx.x * 256 + threadIdx.x;
  float c = 0.0f;
  #pragma unroll
  for (int j = 0; j < CHUNKS; ++j) {
    cin[j * NCH + ch] = c;
    c = cF[j * NCH + ch] * c + cH[j * NCH + ch];
  }
}

__global__ __launch_bounds__(256) void scanC(float* __restrict__ li,
                                             const float* __restrict__ lf,
                                             const float* __restrict__ cin) {
  int gid = blockIdx.x * 256 + threadIdx.x;
  int j = gid >> 12;
  int ch = gid & 4095;
  int b = ch >> 11;
  int hid = ch & 2047;
  size_t base = ((size_t)b * T_ + (size_t)j * CLEN) * HID_ + hid;
  float c = cin[j * NCH + ch];
  for (int t = 0; t < CLEN; ++t) {
    size_t idx = base + (size_t)t * HID_;
    li[idx] = li[idx] + lf[idx] * c;
  }
}


__global__ __launch_bounds__(256) void gate_norm_quant(
    const float* __restrict__ g, const float* __restrict__ o,
    const float* __restrict__ gnW, const float* __restrict__ nO,
    i8* __restrict__ qO, float* __restrict__ sO) {
  __shared__ float red[256];
  const int tid = threadIdx.x;
  const size_t row = blockIdx.x;
  const float* gp = g + row * (size_t)HID_;
  const float* op = o + row * (size_t)HID_;
  float gv[8], ov[8];
  float ssg = 0.0f;
  #pragma unroll
  for (int j = 0; j < 8; ++j) {
    gv[j] = gp[j * 256 + tid];
    ov[j] = op[j * 256 + tid];
    ssg += gv[j] * gv[j];
  }
  float rg = rsqrtf(block_sum(ssg, red) * (1.0f / (float)HID_) + 1e-5f);
  float y[8];
  float ssy = 0.0f;
  #pragma unroll
  for (int j = 0; j < 8; ++j) {
    y[j] = gv[j] * rg * gnW[j * 256 + tid] * ov[j] * sigmf(ov[j]);
    ssy += y[j] * y[j];
  }
  float ry = rsqrtf(block_sum(ssy, red) * (1.0f / (float)HID_) + 1e-8f);
  float z[8];
  float am = 0.0f;
  #pragma unroll
  for (int j = 0; j < 8; ++j) {
    z[j] = y[j] * ry * nO[j * 256 + tid];
    am = fmaxf(am, fabsf(z[j]));
  }
  float amax = fmaxf(block_max(am, red), 1e-5f);
  float s = 127.0f / amax;
  #pragma unroll
  for (int j = 0; j < 8; ++j) {
    float q = fminf(fmaxf(rintf(z[j] * s), -128.0f), 127.0f);
    qO[row * (size_t)HID_ + j * 256 + tid] = (i8)q;
  }
  if (tid == 0) sO[row] = amax * (1.0f / 127.0f);
}


extern "C" void kernel_launch(void* const* d_in, const int* in_sizes, int n_in,
                              void* d_out, int out_size, void* d_ws, size_t ws_size,
                              hipStream_t stream) {
  (void)in_sizes; (void)n_in; (void)out_size; (void)ws_size;

  const float* hs  = (const float*)d_in[0];
  const float* wI  = (const float*)d_in[1];
  const float* wF  = (const float*)d_in[2];
  const float* wG  = (const float*)d_in[3];
  const float* wO  = (const float*)d_in[4];
  const float* nI  = (const float*)d_in[5];
  const float* nF  = (const float*)d_in[6];
  const float* nG  = (const float*)d_in[7];
  const float* nO  = (const float*)d_in[8];
  const float* gnW = (const float*)d_in[9];
  float* out = (float*)d_out;

  char* p = (char*)d_ws;
  size_t off = 0;
  i8* wq[4];
  for (int i = 0; i < 4; ++i) { wq[i] = (i8*)(p + off); off += (size_t)WELEMS; }
  float* wsc = (float*)(p + off); off += 256;
  i8* xq[4];
  for (int i = 0; i < 4; ++i) { xq[i] = (i8*)(p + off); off += (size_t)ROWS_ * HID_; }
  float* asc[4];
  for (int i = 0; i < 4; ++i) { asc[i] = (float*)(p + off); off += (size_t)ROWS_ * 4; }
  float* linI = (float*)(p + off); off += (size_t)ROWS_ * HID_ * 4;
  float* linF = (float*)(p + off); off += (size_t)ROWS_ * HID_ * 4;
  float* linG = (float*)(p + off); off += (size_t)ROWS_ * HID_ * 4;
  float* cF   = (float*)(p + off); off += (size_t)CHUNKS * NCH * 4;
  float* cH   = (float*)(p + off); off += (size_t)CHUNKS * NCH * 4;
  float* cin  = (float*)(p + off); off += (size_t)CHUNKS * NCH * 4;
  float* part = (float*)(p + off); off += 4096;

  const float* wsrc[4] = {wI, wF, wG, wO};

  for (int m = 0; m < 4; ++m) {
    wabs_partial<<<1024, 256, 0, stream>>>(wsrc[m], part);
    wabs_final<<<1, 256, 0, stream>>>(part, wsc + m);
    wquant<<<4096, 256, 0, stream>>>(wsrc[m], wsc + m, wq[m]);
  }

  act_rms_quant3<<<ROWS_, 256, 0, stream>>>(hs, nI, nF, nG,
                                            xq[0], xq[1], xq[2],
                                            asc[0], asc[1], asc[2]);

  dim3 gg(HID_ / BN, ROWS_ / BM);
  gemm_iu8<<<gg, 256, 0, stream>>>(xq[0], wq[0], asc[0], wsc + 0, linI, ROWS_, HID_, HID_);
  gemm_iu8<<<gg, 256, 0, stream>>>(xq[1], wq[1], asc[1], wsc + 1, linF, ROWS_, HID_, HID_);
  gemm_iu8<<<gg, 256, 0, stream>>>(xq[2], wq[2], asc[2], wsc + 2, linG, ROWS_, HID_, HID_);

  size_t nel = (size_t)ROWS_ * HID_;
  gates_ew<<<(unsigned)(nel / 256), 256, 0, stream>>>(linI, linF, nel);

  scanA<<<(NCH * CHUNKS) / 256, 256, 0, stream>>>(linI, linF, cF, cH);
  scanB<<<NCH / 256, 256, 0, stream>>>(cF, cH, cin);
  scanC<<<(NCH * CHUNKS) / 256, 256, 0, stream>>>(linI, linF, cin);

  gate_norm_quant<<<ROWS_, 256, 0, stream>>>(linG, linI, gnW, nO, xq[3], asc[3]);

  gemm_iu8<<<gg, 256, 0, stream>>>(xq[3], wq[3], asc[3], wsc + 3, out, ROWS_, HID_, HID_);
}
